// VisionMamba_target_82197084111576
// MI455X (gfx1250) — matrix operand order repaired and checked by reading, not compiled, not run on hardware
//
#include <hip/hip_runtime.h>

// ---------------- problem constants ----------------
constexpr int NB = 4, NL = 409, NDM = 192, NDI = 384, NDS = 16, NDR = 12, NK = 4, NDEPTH = 24;
constexpr int NM  = NB * NL;           // 1636 token rows
constexpr int MT  = (NM + 15) / 16;    // 103 M tiles
constexpr int MPAD = MT * 16;          // 1648 padded rows
constexpr int NXE = NDR + 2 * NDS;     // 44 x_proj outputs
constexpr int NXP = 48;                // x_proj rows padded to 3 N-tiles
constexpr int NXL = 48;                // padded ld for x_dbl

#define CDIV(a, b) (((a) + (b) - 1) / (b))

typedef __attribute__((ext_vector_type(16))) _Float16 v16h;
typedef __attribute__((ext_vector_type(8)))  _Float16 v8h;
typedef __attribute__((ext_vector_type(8)))  float    v8f;

// ---------------- f32 -> f16 weight conversion ----------------
__global__ void k_f16cvt(const float* __restrict__ src, _Float16* __restrict__ dst, int n) {
    int i = blockIdx.x * 256 + threadIdx.x;
    if (i < n) dst[i] = (_Float16)src[i];
}

// x_proj weights: [DEPTH, 44, NDI] f32 -> [DEPTH, 48, NDI] f16, rows 44..47 zero
__global__ void k_cvt_xproj(const float* __restrict__ src, _Float16* __restrict__ dst) {
    int i = blockIdx.x * 256 + threadIdx.x;
    if (i >= NDEPTH * NXP * NDI) return;
    int c = i % NDI;
    int r = (i / NDI) % NXP;
    int l = i / (NDI * NXP);
    dst[i] = (r < NXE) ? (_Float16)src[((size_t)l * NXE + r) * NDI + c] : (_Float16)0.f;
}

// ---------------- init: hidden = tokens, residual = 0 ----------------
__global__ void k_init(const float* __restrict__ tokens, float* __restrict__ hidden,
                       float* __restrict__ residual, int n) {
    int i = blockIdx.x * 256 + threadIdx.x;
    if (i < n) { hidden[i] = tokens[i]; residual[i] = 0.f; }
}

// ---------------- fused residual-add + RMSNorm -> f16 activations ----------------
__global__ void k_addnorm(const float* __restrict__ hidden, float* __restrict__ residual,
                          const float* __restrict__ nw, _Float16* __restrict__ out16) {
    int wave = threadIdx.x >> 5, lane = threadIdx.x & 31;
    int m = blockIdx.x * 8 + wave;
    if (m >= NM) return;
    float v[6]; float ss = 0.f;
#pragma unroll
    for (int i = 0; i < 6; ++i) {
        int c = lane + 32 * i;
        v[i] = residual[(size_t)m * NDM + c] + hidden[(size_t)m * NDM + c];
        ss += v[i] * v[i];
    }
#pragma unroll
    for (int o = 16; o; o >>= 1) ss += __shfl_xor(ss, o, 32);
    float r = rsqrtf(ss / (float)NDM + 1e-5f);
#pragma unroll
    for (int i = 0; i < 6; ++i) {
        int c = lane + 32 * i;
        residual[(size_t)m * NDM + c] = v[i];
        out16[(size_t)m * NDM + c] = (_Float16)(v[i] * r * nw[c]);
    }
}

// ---------------- final RMSNorm -> f32 output ----------------
__global__ void k_finalnorm(const float* __restrict__ hidden, const float* __restrict__ residual,
                            const float* __restrict__ nw, float* __restrict__ out) {
    int wave = threadIdx.x >> 5, lane = threadIdx.x & 31;
    int m = blockIdx.x * 8 + wave;
    if (m >= NM) return;
    float v[6]; float ss = 0.f;
#pragma unroll
    for (int i = 0; i < 6; ++i) {
        int c = lane + 32 * i;
        v[i] = residual[(size_t)m * NDM + c] + hidden[(size_t)m * NDM + c];
        ss += v[i] * v[i];
    }
#pragma unroll
    for (int o = 16; o; o >>= 1) ss += __shfl_xor(ss, o, 32);
    float r = rsqrtf(ss / (float)NDM + 1e-5f);
#pragma unroll
    for (int i = 0; i < 6; ++i) {
        int c = lane + 32 * i;
        out[(size_t)m * NDM + c] = v[i] * r * nw[c];
    }
}

// ---------------- software-pipelined WMMA inner loop ----------------
// Fragments for step k+32 are loaded before the WMMAs for step k are issued, so
// VMEM/LDS latency hides under matrix-op execution. TWO selects 1 vs 2 N tiles
// statically (no exec-mask churn inside the hot loop).
template <int KDIM, bool TWO>
__device__ __forceinline__ void wmma_loop(const _Float16* __restrict__ sArow, int hi,
                                          const _Float16* __restrict__ w0,
                                          const _Float16* __restrict__ w1,
                                          v8f& c0, v8f& c1) {
    v8h  alo = *(const v8h*)&sArow[16 * hi];
    v8h  ahi = *(const v8h*)&sArow[8 + 16 * hi];
    v16h b0  = *(const v16h*)(w0);
    v16h b1  = {};
    if (TWO) b1 = *(const v16h*)(w1);
#pragma unroll
    for (int kk = 0; kk < KDIM; kk += 32) {
        v16h a = __builtin_shufflevector(alo, ahi, 0, 1, 2, 3, 4, 5, 6, 7,
                                                   8, 9, 10, 11, 12, 13, 14, 15);
        v16h bb0 = b0, bb1 = b1;
        if (kk + 32 < KDIM) {   // compile-time under full unroll
            alo = *(const v8h*)&sArow[kk + 32 + 16 * hi];
            ahi = *(const v8h*)&sArow[kk + 40 + 16 * hi];
            b0  = *(const v16h*)(w0 + kk + 32);
            if (TWO) b1 = *(const v16h*)(w1 + kk + 32);
            if (kk + 64 < KDIM) __builtin_prefetch((const void*)(w0 + kk + 64), 0, 1);
        }
        c0 = __builtin_amdgcn_wmma_f32_16x16x32_f16(false, a, false, bb0,
                                                    (short)0, c0, false, false);
        if (TWO)
            c1 = __builtin_amdgcn_wmma_f32_16x16x32_f16(false, a, false, bb1,
                                                        (short)0, c1, false, false);
    }
}

// ---------------- WMMA GEMM: C[MPAD,N] = A16[MPAD,K] * W16[N,K]^T ----------------
// block = 256 (8 waves). A 16xK tile staged in LDS via b128 copies. Each wave
// owns up to two 16x16 N tiles (A-fragment reuse). All loads/stores unguarded:
// M padded to MPAD, N a multiple of 16 (x_proj weights padded to 48 rows).
template <int KDIM>
__global__ void k_gemm(const _Float16* __restrict__ A, const _Float16* __restrict__ W,
                       float* __restrict__ C, int NT, int ldc) {
    constexpr int LDA = KDIM + 8;                 // halves; row stride 16B-aligned
    __shared__ _Float16 sA[16 * LDA];
    const int tid = threadIdx.x;
    const int lane = tid & 31, wave = tid >> 5;
    const int m0 = blockIdx.x * 16;

    // cooperative stage of the 16xK A tile, 8 halves (16B) per copy
    constexpr int CHUNKS = 16 * KDIM / 8;         // 384 (K=192) or 768 (K=384)
#pragma unroll
    for (int c = tid; c < CHUNKS; c += 256) {
        int r  = c / (KDIM / 8);
        int ko = (c % (KDIM / 8)) * 8;
        *(v8h*)&sA[r * LDA + ko] = *(const v8h*)&A[(size_t)(m0 + r) * KDIM + ko];
    }
    __syncthreads();

    int slot = blockIdx.y * 8 + wave;
    int nt0 = slot * 2, nt1 = nt0 + 1;
    if (nt0 >= NT) return;                        // wave-uniform exit (after barrier)
    bool has1 = nt1 < NT;

    const int row = lane & 15;
    const int hi  = lane >> 4;
    const _Float16* sArow = &sA[row * LDA];
    const _Float16* w0 = W + (size_t)(nt0 * 16 + row) * KDIM + 16 * hi;
    const _Float16* w1 = W + (size_t)(nt1 * 16 + row) * KDIM + 16 * hi;

    v8f c0 = {}; v8f c1 = {};
    if (has1) wmma_loop<KDIM, true >(sArow, hi, w0, w1, c0, c1);
    else      wmma_loop<KDIM, false>(sArow, hi, w0, w1, c0, c1);

    // store D: VGPR r -> row m0+r (lanes 0-15) / m0+r+8 (lanes 16-31), col = lane&15
#pragma unroll
    for (int r = 0; r < 8; ++r) {
        int m = m0 + r + 8 * hi;
        C[(size_t)m * ldc + nt0 * 16 + row] = c0[r];
        if (has1) C[(size_t)m * ldc + nt1 * 16 + row] = c1[r];
    }
}

// ---------------- depthwise causal conv (K=4) + SiLU, both directions ----------------
__global__ void k_conv_silu(const float* __restrict__ xz,
                            const float* __restrict__ cwf, const float* __restrict__ cbf,
                            const float* __restrict__ cwb, const float* __restrict__ cbb,
                            _Float16* __restrict__ xff, _Float16* __restrict__ xfb) {
    int idx = blockIdx.x * 256 + threadIdx.x;
    if (idx >= NM * NDI) return;
    int dir = blockIdx.y;
    const float* cw = dir ? cwb : cwf;
    const float* cb = dir ? cbb : cbf;
    _Float16* out = dir ? xfb : xff;
    int d = idx % NDI, mp = idx / NDI;
    int b = mp / NL, t = mp % NL;
    float s = cb[d];
#pragma unroll
    for (int j = 0; j < NK; ++j) {
        int tj = t - (NK - 1) + j;
        if (tj >= 0) {
            int tok = dir ? (NL - 1 - tj) : tj;
            s += xz[(size_t)(b * NL + tok) * (2 * NDI) + d] * cw[d * NK + j];
        }
    }
    float sig = 1.f / (1.f + __expf(-s));
    out[idx] = (_Float16)(s * sig);
}

// ---------------- dt_proj (K=12) + softplus, both directions ----------------
__global__ void k_dt(const float* __restrict__ xdf, const float* __restrict__ xdb,
                     const float* __restrict__ dtwf, const float* __restrict__ dtbf,
                     const float* __restrict__ dtwb, const float* __restrict__ dtbb,
                     float* __restrict__ dtf, float* __restrict__ dtb_o) {
    int idx = blockIdx.x * 256 + threadIdx.x;
    if (idx >= NM * NDI) return;
    int dir = blockIdx.y;
    const float* xd = dir ? xdb : xdf;
    const float* w  = dir ? dtwb : dtwf;
    const float* bb = dir ? dtbb : dtbf;
    float* o = dir ? dtb_o : dtf;
    int d = idx % NDI, m = idx / NDI;
    float s = bb[d];
#pragma unroll
    for (int r = 0; r < NDR; ++r) s += xd[(size_t)m * NXL + r] * w[d * NDR + r];
    o[idx] = (s > 20.f) ? s : log1pf(__expf(s));
}

// ---------------- selective scan: sequential over L, both directions ----------------
__global__ void k_scan(const _Float16* __restrict__ xff, const _Float16* __restrict__ xfb,
                       const float* __restrict__ dtf, const float* __restrict__ dtb,
                       const float* __restrict__ xdf, const float* __restrict__ xdb,
                       const float* __restrict__ alf, const float* __restrict__ alb,
                       const float* __restrict__ dpf, const float* __restrict__ dpb,
                       float* __restrict__ yf, float* __restrict__ yb) {
    int dir = blockIdx.z;
    const _Float16* xf = dir ? xfb : xff;
    const float* dt = dir ? dtb : dtf;
    const float* xd = dir ? xdb : xdf;
    const float* al = dir ? alb : alf;
    const float* dp = dir ? dpb : dpf;
    float* y = dir ? yb : yf;

    int d = blockIdx.x * 128 + threadIdx.x;   // 0..383
    int b = blockIdx.y;
    float a[NDS], h[NDS];
#pragma unroll
    for (int n = 0; n < NDS; ++n) { a[n] = -__expf(al[d * NDS + n]); h[n] = 0.f; }
    float Dd = dp[d];

    __shared__ float sBC[32];
    for (int t = 0; t < NL; ++t) {
        int m = b * NL + t;
        if (threadIdx.x < 32) sBC[threadIdx.x] = xd[(size_t)m * NXL + NDR + threadIdx.x];
        __syncthreads();
        float u   = (float)xf[(size_t)m * NDI + d];
        float del = dt[(size_t)m * NDI + d];
        float du  = del * u;
        float acc = 0.f;
#pragma unroll
        for (int n = 0; n < NDS; ++n) {
            h[n] = h[n] * __expf(del * a[n]) + du * sBC[n];
            acc += h[n] * sBC[16 + n];
        }
        y[(size_t)m * NDI + d] = acc + u * Dd;
        __syncthreads();
    }
}

// ---------------- gate: (y_f + flip(y_b)) * silu(z) * 0.5 -> f16 ----------------
__global__ void k_gate(const float* __restrict__ yf, const float* __restrict__ yb,
                       const float* __restrict__ xz, _Float16* __restrict__ yg) {
    int idx = blockIdx.x * 256 + threadIdx.x;
    if (idx >= NM * NDI) return;
    int d = idx % NDI, m = idx / NDI;
    int b = m / NL, t = m % NL;
    float z = xz[(size_t)m * (2 * NDI) + NDI + d];
    float yv = yf[idx] + yb[(size_t)(b * NL + (NL - 1 - t)) * NDI + d];
    float g = z / (1.f + __expf(-z));
    yg[idx] = (_Float16)(yv * g * 0.5f);
}

// ---------------- host-side orchestration ----------------
extern "C" void kernel_launch(void* const* d_in, const int* in_sizes, int n_in,
                              void* d_out, int out_size, void* d_ws, size_t ws_size,
                              hipStream_t stream) {
    const float* tokens    = (const float*)d_in[0];
    const float* norm_w    = (const float*)d_in[1];
    const float* in_proj_w = (const float*)d_in[2];
    const float* conv_w    = (const float*)d_in[3];
    const float* conv_b    = (const float*)d_in[4];
    const float* conv_w_b  = (const float*)d_in[5];
    const float* conv_b_b  = (const float*)d_in[6];
    const float* x_proj_w  = (const float*)d_in[7];
    const float* x_proj_wb = (const float*)d_in[8];
    const float* dt_proj_w = (const float*)d_in[9];
    const float* dt_bias   = (const float*)d_in[10];
    const float* dt_proj_wb= (const float*)d_in[11];
    const float* dt_bias_b = (const float*)d_in[12];
    const float* A_log     = (const float*)d_in[13];
    const float* A_log_b   = (const float*)d_in[14];
    const float* D_param   = (const float*)d_in[15];
    const float* D_param_b = (const float*)d_in[16];
    const float* out_proj_w= (const float*)d_in[17];
    const float* norm_f_w  = (const float*)d_in[18];

    // workspace carve (256B aligned blocks)
    size_t off = 0;
    auto carve = [&](size_t elems, size_t esz) -> void* {
        void* p = (char*)d_ws + off;
        off += ((elems * esz + 255) / 256) * 256;
        return p;
    };
    _Float16* w16_in  = (_Float16*)carve((size_t)NDEPTH * 2 * NDI * NDM, 2);
    _Float16* w16_out = (_Float16*)carve((size_t)NDEPTH * NDM * NDI, 2);
    _Float16* w16_xpf = (_Float16*)carve((size_t)NDEPTH * NXP * NDI, 2);
    _Float16* w16_xpb = (_Float16*)carve((size_t)NDEPTH * NXP * NDI, 2);
    _Float16* hn16    = (_Float16*)carve((size_t)MPAD * NDM, 2);
    _Float16* xf16f   = (_Float16*)carve((size_t)MPAD * NDI, 2);
    _Float16* xf16b   = (_Float16*)carve((size_t)MPAD * NDI, 2);
    _Float16* yg16    = (_Float16*)carve((size_t)MPAD * NDI, 2);
    float* residual = (float*)carve((size_t)NM * NDM, 4);
    float* hidden   = (float*)carve((size_t)MPAD * NDM, 4);
    float* xz       = (float*)carve((size_t)MPAD * 2 * NDI, 4);
    float* xdf      = (float*)carve((size_t)MPAD * NXL, 4);
    float* xdb      = (float*)carve((size_t)MPAD * NXL, 4);
    float* dtf      = (float*)carve((size_t)NM * NDI, 4);
    float* dtb      = (float*)carve((size_t)NM * NDI, 4);
    float* yfbuf    = (float*)carve((size_t)NM * NDI, 4);
    float* ybbuf    = (float*)carve((size_t)NM * NDI, 4);

    // one-time weight conversion to f16 (x_proj padded to 48 rows)
    {
        int n;
        n = NDEPTH * 2 * NDI * NDM;
        k_f16cvt<<<CDIV(n, 256), 256, 0, stream>>>(in_proj_w, w16_in, n);
        n = NDEPTH * NDM * NDI;
        k_f16cvt<<<CDIV(n, 256), 256, 0, stream>>>(out_proj_w, w16_out, n);
        n = NDEPTH * NXP * NDI;
        k_cvt_xproj<<<CDIV(n, 256), 256, 0, stream>>>(x_proj_w,  w16_xpf);
        k_cvt_xproj<<<CDIV(n, 256), 256, 0, stream>>>(x_proj_wb, w16_xpb);
    }
    k_init<<<CDIV(NM * NDM, 256), 256, 0, stream>>>(tokens, hidden, residual, NM * NDM);

    for (int l = 0; l < NDEPTH; ++l) {
        k_addnorm<<<CDIV(NM, 8), 256, 0, stream>>>(hidden, residual,
                                                   norm_w + (size_t)l * NDM, hn16);

        // xz = hn16 @ in_proj^T : MPAD x 768, K=192 -> 48 N tiles, grid.y=3
        k_gemm<NDM><<<dim3(MT, 3), 256, 0, stream>>>(
            hn16, w16_in + (size_t)l * 2 * NDI * NDM, xz, 48, 2 * NDI);

        k_conv_silu<<<dim3(CDIV(NM * NDI, 256), 2), 256, 0, stream>>>(
            xz,
            conv_w  + (size_t)l * NDI * NK, conv_b  + (size_t)l * NDI,
            conv_w_b+ (size_t)l * NDI * NK, conv_b_b+ (size_t)l * NDI,
            xf16f, xf16b);

        // x_dbl = xf @ x_proj^T : MPAD x 48, K=384 -> 3 N tiles
        k_gemm<NDI><<<dim3(MT, 1), 256, 0, stream>>>(
            xf16f, w16_xpf + (size_t)l * NXP * NDI, xdf, 3, NXL);
        k_gemm<NDI><<<dim3(MT, 1), 256, 0, stream>>>(
            xf16b, w16_xpb + (size_t)l * NXP * NDI, xdb, 3, NXL);

        k_dt<<<dim3(CDIV(NM * NDI, 256), 2), 256, 0, stream>>>(
            xdf, xdb,
            dt_proj_w + (size_t)l * NDI * NDR, dt_bias  + (size_t)l * NDI,
            dt_proj_wb+ (size_t)l * NDI * NDR, dt_bias_b+ (size_t)l * NDI,
            dtf, dtb);

        k_scan<<<dim3(3, NB, 2), 128, 0, stream>>>(
            xf16f, xf16b, dtf, dtb, xdf, xdb,
            A_log   + (size_t)l * NDI * NDS, A_log_b  + (size_t)l * NDI * NDS,
            D_param + (size_t)l * NDI,       D_param_b+ (size_t)l * NDI,
            yfbuf, ybbuf);

        k_gate<<<CDIV(NM * NDI, 256), 256, 0, stream>>>(yfbuf, ybbuf, xz, yg16);

        // hidden = yg @ out_proj^T : MPAD x 192, K=384 -> 12 N tiles
        k_gemm<NDI><<<dim3(MT, 1), 256, 0, stream>>>(
            yg16, w16_out + (size_t)l * NDM * NDI, hidden, 12, NDM);
    }

    k_finalnorm<<<CDIV(NM, 8), 256, 0, stream>>>(hidden, residual, norm_f_w, (float*)d_out);
}
